// CasualAttention_9474697855267
// MI455X (gfx1250) — compile-verified, matrix operand order checked by reading, not run on hardware
//
#include <hip/hip_runtime.h>
#include <hip/hip_bf16.h>

// ---------------- problem constants ----------------
#define B_   2
#define S_   2048
#define D_   2048
#define H_   16
#define HKV_ 4
#define HD_  128
#define SCALE_ 0.08838834764831843f
#define NEG_  (-1000000000.0f)

typedef __bf16 bf16_t;
typedef __attribute__((ext_vector_type(16))) __bf16 v16bf;
typedef __attribute__((ext_vector_type(8)))  float  v8f;

static __device__ __forceinline__ unsigned rne16(float f) {
    unsigned u = __builtin_bit_cast(unsigned, f);
    return ((u + 0x7FFFu + ((u >> 16) & 1u)) >> 16) & 0xFFFFu;
}
static __device__ __forceinline__ unsigned packbf2(float lo, float hi) {
    return rne16(lo) | (rne16(hi) << 16);
}
static __device__ __forceinline__ bf16_t f2b(float f) {
    unsigned short s = (unsigned short)rne16(f);
    return __builtin_bit_cast(bf16_t, s);
}
static __device__ __forceinline__ float b2f(bf16_t h) {
    unsigned short s = __builtin_bit_cast(unsigned short, h);
    unsigned int u = ((unsigned int)s) << 16;
    return __builtin_bit_cast(float, u);
}

// ---- WMMA fragment loader from dword-packed LDS (2x ds_load_b128 / frag) ----
// CDNA5 16-bit 16x32 layout (ISA 7.12.2): lane<16 -> row=lane, K {0..7,16..23};
// lane>=16 -> row=lane-16, K {8..15,24..31}.  In dwords: u[0..3]=rp[4h+0..3],
// u[4..7]=rp[8+4h+0..3].  strideU must be a multiple of 4 (16B alignment).
union FragU { v16bf v; unsigned u[8]; };
static __device__ __forceinline__ v16bf frag_ld(const unsigned* base, int strideU, int lane) {
    int m = lane & 15, half = lane >> 4;
    const unsigned* rp = base + m * strideU + 4 * half;
    FragU f;
#pragma unroll
    for (int j = 0; j < 4; ++j) f.u[j] = rp[j];
#pragma unroll
    for (int j = 0; j < 4; ++j) f.u[4 + j] = rp[8 + j];
    return f.v;
}
static __device__ __forceinline__ v8f wmma_bf16(v16bf a, v16bf b, v8f c) {
    return __builtin_amdgcn_wmma_f32_16x16x32_bf16(false, a, false, b, (short)0, c, false, false);
}

// =====================================================================
// Kernel 1/6: generic GEMM  C(MxN) = A(Mx2048,f32) x W(2048xN,f32)
// mode 0: plain fp32 row-major out.  mode 1: bf16 head-scatter to
//         (B, HN, S, HD).  Block 256 (8 waves), tile 64x128, BK=32.
// W tile is transposed at staging so B-fragments are row-major-per-lane.
// =====================================================================
__global__ void k_gemm(const float* __restrict__ A, const float* __restrict__ W,
                       int N, void* __restrict__ outp, int mode, int HN)
{
    __shared__ __align__(16) unsigned Asu[64 * 20];   // 64 rows x 32 bf16 (16 dw + 4 pad)
    __shared__ __align__(16) bf16_t   Bst[128 * 40];  // transposed: 128 n-rows x 32 k (+8 pad)
    const unsigned* Bstu = (const unsigned*)Bst;
    const int t = threadIdx.x;
    const int lane = t & 31, w = t >> 5;
    const int wm = w >> 2, wn = w & 3;
    const int mbase = blockIdx.x * 64, nbase = blockIdx.y * 128;

    v8f z = {0.f,0.f,0.f,0.f,0.f,0.f,0.f,0.f};
    v8f acc[2][2] = {{z, z}, {z, z}};

    for (int k0 = 0; k0 < 2048; k0 += 32) {
        __syncthreads();
#pragma unroll
        for (int i = 0; i < 2; ++i) {                     // stage A 64x32 (packed dwords)
            int idx = t + i * 256;
            int row = idx >> 3, c4 = (idx & 7) * 4;
            float4 v = *(const float4*)(A + (size_t)(mbase + row) * 2048 + k0 + c4);
            Asu[row * 20 + (c4 >> 1) + 0] = packbf2(v.x, v.y);
            Asu[row * 20 + (c4 >> 1) + 1] = packbf2(v.z, v.w);
        }
#pragma unroll
        for (int i = 0; i < 4; ++i) {                     // stage W 32x128 -> transposed
            int idx = t + i * 256;
            int row = idx >> 5, c4 = (idx & 31) * 4;
            float4 v = *(const float4*)(W + (size_t)(k0 + row) * N + nbase + c4);
            Bst[(c4 + 0) * 40 + row] = f2b(v.x);
            Bst[(c4 + 1) * 40 + row] = f2b(v.y);
            Bst[(c4 + 2) * 40 + row] = f2b(v.z);
            Bst[(c4 + 3) * 40 + row] = f2b(v.w);
        }
        __syncthreads();
        v16bf a0 = frag_ld(Asu + (wm * 32) * 20, 20, lane);
        v16bf a1 = frag_ld(Asu + (wm * 32 + 16) * 20, 20, lane);
        v16bf b0 = frag_ld(Bstu + (wn * 32) * 20, 20, lane);
        v16bf b1 = frag_ld(Bstu + (wn * 32 + 16) * 20, 20, lane);
        acc[0][0] = wmma_bf16(a0, b0, acc[0][0]);
        acc[0][1] = wmma_bf16(a0, b1, acc[0][1]);
        acc[1][0] = wmma_bf16(a1, b0, acc[1][0]);
        acc[1][1] = wmma_bf16(a1, b1, acc[1][1]);
    }

    const int half = lane >> 4, nl = lane & 15;
#pragma unroll
    for (int i = 0; i < 2; ++i)
#pragma unroll
        for (int j = 0; j < 2; ++j)
#pragma unroll
            for (int r = 0; r < 8; ++r) {
                int m = mbase + wm * 32 + i * 16 + r + 8 * half;
                int n = nbase + wn * 32 + j * 16 + nl;
                float v = acc[i][j][r];
                if (mode == 0) {
                    ((float*)outp)[(size_t)m * N + n] = v;
                } else {
                    int bb = m >> 11, ss = m & 2047, hh = n >> 7, dd = n & 127;
                    ((bf16_t*)outp)[(((size_t)(bb * HN + hh) * 2048) + ss) * 128 + dd] = f2b(v);
                }
            }
}

// =====================================================================
// Kernel 2: RoPE in place on bf16 (B, NH, S, HD).
// =====================================================================
__global__ void k_rope(bf16_t* __restrict__ p, const float* __restrict__ cosp,
                       const float* __restrict__ sinp, int NH)
{
    int idx = blockIdx.x * 256 + threadIdx.x;
    int total = B_ * NH * 2048 * 64;
    if (idx >= total) return;
    int d = idx & 63;
    int tmp = idx >> 6;
    int s = tmp & 2047; tmp >>= 11;
    int h = tmp % NH;
    int b = tmp / NH;
    size_t base = (((size_t)(b * NH + h) * 2048) + s) * 128;
    size_t cbase = ((size_t)b * 2048 + s) * 128 + d;
    float c = cosp[cbase], sn = sinp[cbase];
    float x1 = b2f(p[base + d]);
    float x2 = b2f(p[base + d + 64]);
    p[base + d]      = f2b(x1 * c - x2 * sn);
    p[base + d + 64] = f2b(x2 * c + x1 * sn);
}

// =====================================================================
// Kernel 3: raw masked scores.  Grid (qt=16, h=16, b=2), block 256.
// Q tile LDS-resident; K tiles streamed via global_load_async_to_lds_b128
// (CDNA5 async-LDS DMA, ASYNCcnt).  Writes raw scaled+masked scores fp32.
// =====================================================================
__global__ void k_scores(const bf16_t* __restrict__ qb, const bf16_t* __restrict__ kb,
                         float* __restrict__ attn)
{
    __shared__ __align__(16) unsigned Qsu[128 * 68];  // 128 x 128 bf16, stride 68 dw
    __shared__ __align__(16) unsigned Ksu[64 * 68];   // 64 x 128 bf16
    const int qt = blockIdx.x, h = blockIdx.y, b = blockIdx.z;
    const int hk = h >> 2;
    const int t = threadIdx.x, lane = t & 31, w = t >> 5;
    const int wm = w >> 1, wn = w & 1;     // 4 x 2 waves over 128q x 64k

    const uint4* qsrc = (const uint4*)(qb + (((size_t)(b * H_ + h) * 2048) + (size_t)qt * 128) * 128);
#pragma unroll
    for (int i = 0; i < 8; ++i) {
        int idx = t + 256 * i;
        int row = idx >> 4, c = idx & 15;
        *(uint4*)(Qsu + row * 68 + c * 4) = qsrc[idx];
    }

    const unsigned ldsK = (unsigned)(uintptr_t)Ksu;
    const int ktmax = 2 * qt + 1;
    for (int kt = 0; kt <= ktmax; ++kt) {
        const bf16_t* ksrc = kb + (((size_t)(b * HKV_ + hk) * 2048) + (size_t)kt * 64) * 128;
        __syncthreads();
        // ---- CDNA5 async DMA: global -> LDS, tracked by ASYNCcnt ----
#pragma unroll
        for (int i = 0; i < 4; ++i) {
            int idx = t + 256 * i;
            int key = idx >> 4, c = idx & 15;
            unsigned lds = ldsK + (unsigned)(key * 272 + c * 16);
            const void* gp = (const void*)(ksrc + (size_t)key * 128 + c * 8);
            asm volatile("global_load_async_to_lds_b128 %0, %1, off"
                         :: "v"(lds), "v"(gp) : "memory");
        }
        asm volatile("s_wait_asynccnt 0" ::: "memory");
        __syncthreads();

        v8f z = {0.f,0.f,0.f,0.f,0.f,0.f,0.f,0.f};
        v8f acc[2][2] = {{z, z}, {z, z}};
#pragma unroll
        for (int kk = 0; kk < 4; ++kk) {
            int du = kk * 16;   // 32 bf16 = 16 dwords per k-step
            v16bf a0 = frag_ld(Qsu + (wm * 32) * 68 + du, 68, lane);
            v16bf a1 = frag_ld(Qsu + (wm * 32 + 16) * 68 + du, 68, lane);
            v16bf b0 = frag_ld(Ksu + (wn * 32) * 68 + du, 68, lane);
            v16bf b1 = frag_ld(Ksu + (wn * 32 + 16) * 68 + du, 68, lane);
            acc[0][0] = wmma_bf16(a0, b0, acc[0][0]);
            acc[0][1] = wmma_bf16(a0, b1, acc[0][1]);
            acc[1][0] = wmma_bf16(a1, b0, acc[1][0]);
            acc[1][1] = wmma_bf16(a1, b1, acc[1][1]);
        }

        const int half = lane >> 4, nl = lane & 15;
#pragma unroll
        for (int i = 0; i < 2; ++i)
#pragma unroll
            for (int j = 0; j < 2; ++j)
#pragma unroll
                for (int r = 0; r < 8; ++r) {
                    int q  = qt * 128 + wm * 32 + i * 16 + r + 8 * half;
                    int kg = kt * 64 + wn * 32 + j * 16 + nl;
                    float v = acc[i][j][r] * SCALE_ + ((kg <= q) ? 0.0f : NEG_);
                    attn[(((size_t)(b * H_ + h) * 2048) + q) * 2048 + kg] = v;
                }
    }
}

// =====================================================================
// Kernel 4: per-row softmax stats over the causally-valid prefix.
// =====================================================================
__global__ void k_rowstats(const float* __restrict__ attn,
                           float* __restrict__ rowm, float* __restrict__ rowl)
{
    int row = blockIdx.x * 8 + (threadIdx.x >> 5);
    int lane = threadIdx.x & 31;
    int q = row & 2047;
    int len = ((q >> 7) + 1) * 128;
    const float* rp = attn + (size_t)row * 2048;
    float mx = -3.4e38f;
    for (int k = lane; k < len; k += 32) mx = fmaxf(mx, rp[k]);
#pragma unroll
    for (int off = 16; off > 0; off >>= 1) mx = fmaxf(mx, __shfl_xor(mx, off, 32));
    float sm = 0.f;
    for (int k = lane; k < len; k += 32) sm += __expf(rp[k] - mx);
#pragma unroll
    for (int off = 16; off > 0; off >>= 1) sm += __shfl_xor(sm, off, 32);
    if (lane == 0) { rowm[row] = mx; rowl[row] = sm; }
}

// =====================================================================
// Kernel 5: normalize attn in place + P @ V.  Grid (qt,h,b), block 256.
// V tile transposed at staging so both WMMA operands are row-major frags.
// =====================================================================
__global__ void k_pv(const bf16_t* __restrict__ vb, float* __restrict__ attn,
                     const float* __restrict__ rowm, const float* __restrict__ rowl,
                     float* __restrict__ outpre)
{
    __shared__ __align__(16) unsigned Psu[128 * 36];  // 128 q-rows x 64 bf16 (32 dw + 4 pad)
    __shared__ __align__(16) bf16_t   Vtb[128 * 72];  // transposed: 128 d-rows x 64 keys (+8)
    __shared__ float smx[128];
    __shared__ float ssum[128];
    const unsigned* Vtu = (const unsigned*)Vtb;
    const int qt = blockIdx.x, h = blockIdx.y, b = blockIdx.z;
    const int hk = h >> 2;
    const int t = threadIdx.x, lane = t & 31, w = t >> 5;
    const int wm = w >> 1, wn = w & 1;     // 4 x 2 waves over 128q x 128d

    if (t < 128) {
        int row = (b * H_ + h) * 2048 + qt * 128 + t;
        smx[t]  = rowm[row];
        ssum[t] = rowl[row];
    }

    v8f z = {0.f,0.f,0.f,0.f,0.f,0.f,0.f,0.f};
    v8f acc[2][4] = {{z, z, z, z}, {z, z, z, z}};
    const int ktmax = 2 * qt + 1;

    for (int kt = 0; kt < 32; ++kt) {
        float* ab = attn + (((size_t)(b * H_ + h) * 2048) + (size_t)qt * 128) * 2048 + (size_t)kt * 64;
        bool valid = (kt <= ktmax);
        __syncthreads();
        if (valid) {
            // normalize pairwise, write probs back, pack into LDS dwords
#pragma unroll
            for (int i = 0; i < 16; ++i) {
                int e = 2 * t + 512 * i;
                int r = e >> 6, c = e & 63;
                float2 sv = *(float2*)(ab + (size_t)r * 2048 + c);
                float p0 = __expf(sv.x - smx[r]) / ssum[r];
                float p1 = __expf(sv.y - smx[r]) / ssum[r];
                *(float2*)(ab + (size_t)r * 2048 + c) = make_float2(p0, p1);
                Psu[r * 36 + (c >> 1)] = packbf2(p0, p1);
            }
            // stage V transposed: Vtb[d][key]
            const uint4* vsrc = (const uint4*)(vb + (((size_t)(b * HKV_ + hk) * 2048) + (size_t)kt * 64) * 128);
#pragma unroll
            for (int i = 0; i < 4; ++i) {
                int idx = t + 256 * i;
                int key = idx >> 4, dbase = (idx & 15) * 8;
                uint4 v = vsrc[idx];
                unsigned arr[4] = {v.x, v.y, v.z, v.w};
#pragma unroll
                for (int q2 = 0; q2 < 4; ++q2) {
                    unsigned u = arr[q2];
                    Vtb[(dbase + 2 * q2 + 0) * 72 + key] =
                        __builtin_bit_cast(bf16_t, (unsigned short)(u & 0xFFFFu));
                    Vtb[(dbase + 2 * q2 + 1) * 72 + key] =
                        __builtin_bit_cast(bf16_t, (unsigned short)(u >> 16));
                }
            }
        } else {
#pragma unroll
            for (int i = 0; i < 8; ++i) {
                int e = 4 * t + 1024 * i;
                int r = e >> 6, c = e & 63;
                *(float4*)(ab + (size_t)r * 2048 + c) = make_float4(0.f, 0.f, 0.f, 0.f);
            }
        }
        __syncthreads();
        if (valid) {
#pragma unroll
            for (int kk = 0; kk < 2; ++kk) {
                int ku = kk * 16;   // 32 keys = 16 dwords
                v16bf a0 = frag_ld(Psu + (wm * 32) * 36 + ku, 36, lane);
                v16bf a1 = frag_ld(Psu + (wm * 32 + 16) * 36 + ku, 36, lane);
#pragma unroll
                for (int j = 0; j < 4; ++j) {
                    v16bf bj = frag_ld(Vtu + (wn * 64 + j * 16) * 36 + ku, 36, lane);
                    acc[0][j] = wmma_bf16(a0, bj, acc[0][j]);
                    acc[1][j] = wmma_bf16(a1, bj, acc[1][j]);
                }
            }
        }
    }

    const int half = lane >> 4, nl = lane & 15;
#pragma unroll
    for (int i = 0; i < 2; ++i)
#pragma unroll
        for (int j = 0; j < 4; ++j)
#pragma unroll
            for (int r = 0; r < 8; ++r) {
                int q = qt * 128 + wm * 32 + i * 16 + r + 8 * half;
                int d = wn * 64 + j * 16 + nl;
                outpre[((size_t)b * 2048 + q) * 2048 + h * 128 + d] = acc[i][j][r];
            }
}

// =====================================================================
extern "C" void kernel_launch(void* const* d_in, const int* in_sizes, int n_in,
                              void* d_out, int out_size, void* d_ws, size_t ws_size,
                              hipStream_t stream) {
    (void)in_sizes; (void)n_in; (void)out_size; (void)ws_size;
    const float* x    = (const float*)d_in[0];
    const float* cosp = (const float*)d_in[1];
    const float* sinp = (const float*)d_in[2];
    // d_in[3] = attention_mask (exact causal; computed analytically)
    const float* Wq   = (const float*)d_in[4];
    const float* Wk   = (const float*)d_in[5];
    const float* Wv   = (const float*)d_in[6];
    const float* Wo   = (const float*)d_in[7];

    char* ws = (char*)d_ws;
    bf16_t* qbuf = (bf16_t*)(ws);                        // 16 MiB
    bf16_t* kbuf = (bf16_t*)(ws + 16777216);             //  4 MiB
    bf16_t* vbuf = (bf16_t*)(ws + 20971520);             //  4 MiB
    float*  opre = (float*) (ws + 25165824);             // 32 MiB
    float*  rowm = (float*) (ws + 58720256);             // 256 KiB
    float*  rowl = (float*) (ws + 58982400);             // 256 KiB

    float* outp = (float*)d_out;                         // (B,S,D) fp32
    float* attn = outp + (size_t)B_ * S_ * D_;           // (B,H,S,S) fp32

    dim3 blk(256);
    k_gemm<<<dim3(64, 16), blk, 0, stream>>>(x, Wq, 2048, (void*)qbuf, 1, H_);
    k_gemm<<<dim3(64, 4),  blk, 0, stream>>>(x, Wk, 512,  (void*)kbuf, 1, HKV_);
    k_gemm<<<dim3(64, 4),  blk, 0, stream>>>(x, Wv, 512,  (void*)vbuf, 1, HKV_);
    k_rope<<<dim3(16384), blk, 0, stream>>>(qbuf, cosp, sinp, H_);
    k_rope<<<dim3(4096),  blk, 0, stream>>>(kbuf, cosp, sinp, HKV_);
    k_scores<<<dim3(16, 16, 2), blk, 0, stream>>>(qbuf, kbuf, attn);
    k_rowstats<<<dim3(8192), blk, 0, stream>>>(attn, rowm, rowl);
    k_pv<<<dim3(16, 16, 2), blk, 0, stream>>>(vbuf, attn, rowm, rowl, opre);
    k_gemm<<<dim3(64, 16), blk, 0, stream>>>(opre, Wo, 2048, (void*)outp, 0, 0);
}
